// SSM2DRefiner_2173253452157
// MI455X (gfx1250) — matrix operand order repaired and checked by reading, not compiled, not run on hardware
//
#include <hip/hip_runtime.h>
#include <hip/hip_bf16.h>

// ---- problem sizes (fixed by reference) ----
#define NB 8
#define CCH 256
#define LL 4096
#define PP 256
#define NROWS (NB * LL)      // 32768
#define CHUNK 128
#define NCHUNK (LL / CHUNK)  // 32

typedef __attribute__((ext_vector_type(16))) __bf16 v16bf;
typedef __attribute__((ext_vector_type(8)))  __bf16 v8bf;
typedef __attribute__((ext_vector_type(8)))  float  v8f;

__device__ __forceinline__ v8f zero_v8f() {
  v8f z = {0.f, 0.f, 0.f, 0.f, 0.f, 0.f, 0.f, 0.f};
  return z;
}
__device__ __forceinline__ float2 cmul(float2 a, float2 b) {
  return make_float2(a.x * b.x - a.y * b.y, a.x * b.y + a.y * b.x);
}
// a*x + u (complex fma)
__device__ __forceinline__ float2 cmadd(float2 a, float2 x, float2 u) {
  return make_float2(a.x * x.x - a.y * x.y + u.x, a.x * x.y + a.y * x.x + u.y);
}

// ---- WMMA fragment loaders per CDNA5 ISA VGPR layouts (CDNA5 ISA 7.12.2) ----
// A 16x32 bf16, row-major source, ld in elements.
// lane m=lane&15 holds row m; half=lane>>4 selects K in [h*8,h*8+8) and [16+h*8,...)
__device__ __forceinline__ v16bf load_a16x32(const __bf16* p, int ld, int lane) {
  int m = lane & 15, h = lane >> 4;
  const __bf16* r = p + m * ld + h * 8;
  v8bf lo = *(const v8bf*)r;        // K = h*8 .. h*8+7
  v8bf hi = *(const v8bf*)(r + 16); // K = 16+h*8 ..
  v16bf o;
#pragma unroll
  for (int i = 0; i < 8; i++) { o[i] = lo[i]; o[8 + i] = hi[i]; }
  return o;
}
// B 32x16 bf16 built from weight W (row-major, rows are output cols): B[k][n]=W[n][k].
// lane n=lane&15 -> column n; half selects K in [h*8,h*8+8) and [16+h*8,...)
__device__ __forceinline__ v16bf load_b32x16(const __bf16* p, int ld, int lane) {
  int n = lane & 15, h = lane >> 4;
  const __bf16* r = p + n * ld + h * 8;
  v8bf lo = *(const v8bf*)r;
  v8bf hi = *(const v8bf*)(r + 16);
  v16bf o;
#pragma unroll
  for (int i = 0; i < 8; i++) { o[i] = lo[i]; o[8 + i] = hi[i]; }
  return o;
}
__device__ __forceinline__ v8f wmma_bf16(v16bf a, v16bf b, v8f c) {
  return __builtin_amdgcn_wmma_f32_16x16x32_bf16(false, a, false, b, (short)0, c,
                                                 false, false);
}

// ---- prep1: per-P discretization: Lam_bar, (Lam_bar-1)/Lam, Lam_bar^CHUNK ----
__global__ void k_prep1(const float* __restrict__ lre_, const float* __restrict__ lim_,
                        const float* __restrict__ lstep, float2* __restrict__ LamBar,
                        float2* __restrict__ Coef, float2* __restrict__ LamPow) {
  int p = blockIdx.x * blockDim.x + threadIdx.x;
  if (p >= PP) return;
  float dt = expf(lstep[p]);
  float lre = lre_[p], lim = lim_[p];
  float er = expf(lre * dt);
  float2 lb = make_float2(er * cosf(lim * dt), er * sinf(lim * dt));
  float2 num = make_float2(lb.x - 1.f, lb.y);
  float den = lre * lre + lim * lim;
  float2 cf = make_float2((num.x * lre + num.y * lim) / den,
                          (num.y * lre - num.x * lim) / den);
  LamBar[p] = lb;
  Coef[p] = cf;
  float2 pw = lb;
#pragma unroll
  for (int i = 0; i < 7; i++) pw = cmul(pw, pw);  // ^128 = ^(2^7)
  LamPow[p] = pw;
}

// ---- prep2: B_bar = coef*B (complex) -> bf16 planes; weight bf16 conversions ----
__global__ void k_prep2(const float* __restrict__ Bre, const float* __restrict__ Bim,
                        const float2* __restrict__ Coef, const float* __restrict__ Cre,
                        const float* __restrict__ Cim, const float* __restrict__ enc,
                        const float* __restrict__ dec, __bf16* __restrict__ Bbre16,
                        __bf16* __restrict__ Bbim16, __bf16* __restrict__ Cre16,
                        __bf16* __restrict__ nCim16, __bf16* __restrict__ enc16,
                        __bf16* __restrict__ dec16) {
  int i = blockIdx.x * blockDim.x + threadIdx.x;
  if (i >= PP * CCH) return;
  int p = i >> 8;
  float2 cf = Coef[p];
  float br = Bre[i], bi = Bim[i];
  Bbre16[i] = (__bf16)(cf.x * br - cf.y * bi);
  Bbim16[i] = (__bf16)(cf.x * bi + cf.y * br);
  Cre16[i] = (__bf16)Cre[i];
  nCim16[i] = (__bf16)(-Cim[i]);  // pre-negate so GEMM2 only accumulates
  enc16[i] = (__bf16)enc[i];
  dec16[i] = (__bf16)dec[i];
}

// ---- LN1: transpose (b,c,L)->(row=b*L+l, c) + layernorm; emit f32 + bf16 ----
__global__ void __launch_bounds__(256) k_ln1(const float* __restrict__ x,
                                             const float* __restrict__ w,
                                             const float* __restrict__ bia,
                                             float* __restrict__ fx32,
                                             __bf16* __restrict__ fx16) {
  __shared__ __bf16 tile[64 * 260];  // padded stride vs banks
  __shared__ float psum[256], psq[256];
  __shared__ float meanA[64], rstdA[64];
  int bb = blockIdx.x >> 6;
  int l0 = (blockIdx.x & 63) << 6;
  int t = threadIdx.x;
  int lt = t & 63, cq = t >> 6;
  float s = 0.f, ss = 0.f;
  const float* xp = x + ((size_t)bb * CCH) * LL + l0 + lt;
#pragma unroll 4
  for (int j = 0; j < 64; j++) {
    int c = cq * 64 + j;
    float v = xp[(size_t)c * LL];  // coalesced along l
    tile[lt * 260 + c] = (__bf16)v;
    s += v;
    ss += v * v;
  }
  psum[t] = s;
  psq[t] = ss;
  __syncthreads();
  if (t < 64) {
    float su = psum[t] + psum[64 + t] + psum[128 + t] + psum[192 + t];
    float sq = psq[t] + psq[64 + t] + psq[128 + t] + psq[192 + t];
    float mean = su * (1.f / 256.f);
    float var = sq * (1.f / 256.f) - mean * mean;
    meanA[t] = mean;
    rstdA[t] = rsqrtf(var + 1e-5f);
  }
  __syncthreads();
  int c = t;
  float wc = w[c], bc = bia[c];
#pragma unroll 4
  for (int l = 0; l < 64; l++) {
    float v = (float)tile[l * 260 + c];
    float o = (v - meanA[l]) * rstdA[l] * wc + bc;
    size_t idx = ((size_t)(bb * LL + l0 + l)) * CCH + c;  // coalesced along c
    fx32[idx] = o;
    fx16[idx] = (__bf16)o;
  }
}

// ---- GEMM1: Bu(complex) = fx(bf16) @ Bbar^T ; wave tile 16x64, re+im accums ----
__global__ void __launch_bounds__(256) k_gemm1(const __bf16* __restrict__ A,
                                               const __bf16* __restrict__ Wre,
                                               const __bf16* __restrict__ Wim,
                                               float2* __restrict__ Bu) {
  int wid = (blockIdx.x * blockDim.x + threadIdx.x) >> 5;
  int lane = threadIdx.x & 31;
  const int NT = PP / 64;  // 4
  int mt = wid / NT, nt = wid - mt * NT;
  if (mt >= NROWS / 16) return;
  v8f accR[4], accI[4];
#pragma unroll
  for (int j = 0; j < 4; j++) { accR[j] = zero_v8f(); accI[j] = zero_v8f(); }
  const __bf16* Ap = A + (size_t)mt * 16 * CCH;
#pragma unroll
  for (int k0 = 0; k0 < CCH; k0 += 32) {
    v16bf a = load_a16x32(Ap + k0, CCH, lane);
#pragma unroll
    for (int j = 0; j < 4; j++) {
      v16bf br = load_b32x16(Wre + (size_t)(nt * 64 + j * 16) * CCH + k0, CCH, lane);
      accR[j] = wmma_bf16(a, br, accR[j]);
      v16bf bi = load_b32x16(Wim + (size_t)(nt * 64 + j * 16) * CCH + k0, CCH, lane);
      accI[j] = wmma_bf16(a, bi, accI[j]);
    }
  }
  int rbase = mt * 16 + ((lane >> 4) << 3);
  int cbase = nt * 64 + (lane & 15);
#pragma unroll
  for (int j = 0; j < 4; j++)
#pragma unroll
    for (int r = 0; r < 8; r++)
      Bu[(size_t)(rbase + r) * PP + cbase + j * 16] =
          make_float2(accR[j][r], accI[j][r]);
}

// ---- scan pass 1: per (b,p,chunk) chunk-end state ----
__global__ void k_scan1(const float2* __restrict__ Bu, const float2* __restrict__ LamBar,
                        float2* __restrict__ carry) {
  int tid = blockIdx.x * blockDim.x + threadIdx.x;
  if (tid >= NB * NCHUNK * PP) return;
  int p = tid & 255;
  int chunk = (tid >> 8) & (NCHUNK - 1);
  int bb = tid >> 13;
  float2 lam = LamBar[p];
  float2 s = make_float2(0.f, 0.f);
  const float2* bp = Bu + ((size_t)(bb * LL + chunk * CHUNK)) * PP + p;
  for (int i = 0; i < CHUNK; i++) s = cmadd(lam, s, bp[(size_t)i * PP]);
  carry[tid] = s;
}

// ---- scan pass 2: exclusive scan of chunk states with factor Lam^CHUNK ----
__global__ void k_scan2(const float2* __restrict__ LamPow, float2* __restrict__ carry) {
  int tid = blockIdx.x * blockDim.x + threadIdx.x;
  if (tid >= NB * PP) return;
  int p = tid & 255, bb = tid >> 8;
  float2 lp = LamPow[p];
  float2 c = make_float2(0.f, 0.f);
  for (int j = 0; j < NCHUNK; j++) {
    size_t idx = ((size_t)(bb * NCHUNK + j)) * PP + p;
    float2 s = carry[idx];
    carry[idx] = c;  // carry-in for chunk j
    c = cmadd(lp, c, s);
  }
}

// ---- scan pass 3: recompute local scan seeded by carry-in; emit xs bf16 planes ----
__global__ void k_scan3(const float2* __restrict__ Bu, const float2* __restrict__ LamBar,
                        const float2* __restrict__ carry, __bf16* __restrict__ xsre,
                        __bf16* __restrict__ xsim) {
  int tid = blockIdx.x * blockDim.x + threadIdx.x;
  if (tid >= NB * NCHUNK * PP) return;
  int p = tid & 255;
  int chunk = (tid >> 8) & (NCHUNK - 1);
  int bb = tid >> 13;
  float2 lam = LamBar[p];
  float2 xv = carry[tid];
  size_t rowbase = (size_t)(bb * LL + chunk * CHUNK);
  const float2* bp = Bu + rowbase * PP + p;
  for (int i = 0; i < CHUNK; i++) {
    xv = cmadd(lam, xv, bp[(size_t)i * PP]);
    size_t o = (rowbase + i) * PP + p;
    xsre[o] = (__bf16)xv.x;
    xsim[o] = (__bf16)xv.y;
  }
}

// ---- GEMM2: t = gelu(2*(xs_re@Cre^T + xs_im@(-Cim)^T) + u*D) + u ; in-place ----
__global__ void __launch_bounds__(256) k_gemm2(const __bf16* __restrict__ Are,
                                               const __bf16* __restrict__ Aim,
                                               const __bf16* __restrict__ Wre,
                                               const __bf16* __restrict__ Wni,
                                               const float* __restrict__ Dv,
                                               float* __restrict__ tio) {
  int wid = (blockIdx.x * blockDim.x + threadIdx.x) >> 5;
  int lane = threadIdx.x & 31;
  const int NT = CCH / 64;
  int mt = wid / NT, nt = wid - mt * NT;
  if (mt >= NROWS / 16) return;
  v8f acc[4];
#pragma unroll
  for (int j = 0; j < 4; j++) acc[j] = zero_v8f();
  size_t abase = (size_t)mt * 16 * PP;
#pragma unroll
  for (int k0 = 0; k0 < PP; k0 += 32) {
    v16bf ar = load_a16x32(Are + abase + k0, PP, lane);
    v16bf ai = load_a16x32(Aim + abase + k0, PP, lane);
#pragma unroll
    for (int j = 0; j < 4; j++) {
      v16bf br = load_b32x16(Wre + (size_t)(nt * 64 + j * 16) * PP + k0, PP, lane);
      acc[j] = wmma_bf16(ar, br, acc[j]);
      v16bf bi = load_b32x16(Wni + (size_t)(nt * 64 + j * 16) * PP + k0, PP, lane);
      acc[j] = wmma_bf16(ai, bi, acc[j]);
    }
  }
  int rbase = mt * 16 + ((lane >> 4) << 3);
  int cbase = nt * 64 + (lane & 15);
#pragma unroll
  for (int j = 0; j < 4; j++)
#pragma unroll
    for (int r = 0; r < 8; r++) {
      int col = cbase + j * 16;
      size_t idx = (size_t)(rbase + r) * CCH + col;
      float u = tio[idx];
      float y = 2.f * acc[j][r] + u * Dv[col];
      float g = 0.5f * y *
                (1.f + tanhf(0.7978845608028654f * (y + 0.044715f * y * y * y)));
      tio[idx] = g + u;  // gelu(y) + residual(fx)
    }
}

// ---- LN2: contiguous rows, one wave per row ----
__global__ void __launch_bounds__(256) k_ln2(const float* __restrict__ tin,
                                             const float* __restrict__ w,
                                             const float* __restrict__ bia,
                                             float* __restrict__ fx2,
                                             __bf16* __restrict__ fx216) {
  int wave = threadIdx.x >> 5, lane = threadIdx.x & 31;
  int row = blockIdx.x * 8 + wave;
  if (row >= NROWS) return;
  const float* rp = tin + (size_t)row * CCH;
  float v[8];
  float s = 0.f, ss = 0.f;
#pragma unroll
  for (int k = 0; k < 8; k++) {
    v[k] = rp[lane + k * 32];
    s += v[k];
    ss += v[k] * v[k];
  }
#pragma unroll
  for (int off = 16; off > 0; off >>= 1) {
    s += __shfl_xor(s, off, 32);
    ss += __shfl_xor(ss, off, 32);
  }
  float mean = s * (1.f / 256.f);
  float rstd = rsqrtf(ss * (1.f / 256.f) - mean * mean + 1e-5f);
  float* op = fx2 + (size_t)row * CCH;
  __bf16* op16 = fx216 + (size_t)row * CCH;
#pragma unroll
  for (int k = 0; k < 8; k++) {
    int c = lane + k * 32;
    float o = (v[k] - mean) * rstd * w[c] + bia[c];
    op[c] = o;
    op16[c] = (__bf16)o;
  }
}

// ---- FF: out = (fx2@enc^T)@dec^T + fx2, fused via LDS bf16 intermediate;
//      transposed store back to (b,c,l). Block tile: 64 rows x 256 cols ----
__global__ void __launch_bounds__(256) k_ff(const __bf16* __restrict__ X16,
                                            const float* __restrict__ X32,
                                            const __bf16* __restrict__ E16,
                                            const __bf16* __restrict__ Dc16,
                                            float* __restrict__ out) {
  __shared__ __align__(32) char smem[64 * 260 / 2 * 4];  // 33280B union
  __bf16* inter = (__bf16*)smem;                         // 64x256 bf16 (32768B)
  float* fbuf = (float*)smem;                            // 32x260 f32 (33280B)
  int t = threadIdx.x, lane = t & 31, w = t >> 5;
  int row0 = blockIdx.x * 64;
  int bb = row0 >> 12;
  int l0 = row0 & (LL - 1);
  int rg = w >> 1;  // 0..3 -> 16-row group
  int ch = w & 1;   // 0..1 -> 128-col half
  // GEMM-a: inter = fx2 @ enc^T
  v8f acc[8];
#pragma unroll
  for (int j = 0; j < 8; j++) acc[j] = zero_v8f();
  const __bf16* Ap = X16 + (size_t)(row0 + rg * 16) * CCH;
#pragma unroll
  for (int k0 = 0; k0 < CCH; k0 += 32) {
    v16bf a = load_a16x32(Ap + k0, CCH, lane);
#pragma unroll
    for (int j = 0; j < 8; j++) {
      v16bf b = load_b32x16(E16 + (size_t)(ch * 128 + j * 16) * CCH + k0, CCH, lane);
      acc[j] = wmma_bf16(a, b, acc[j]);
    }
  }
  int rb = rg * 16 + ((lane >> 4) << 3);
  int cb0 = ch * 128 + (lane & 15);
#pragma unroll
  for (int j = 0; j < 8; j++)
#pragma unroll
    for (int r = 0; r < 8; r++)
      inter[(rb + r) * 256 + cb0 + j * 16] = (__bf16)acc[j][r];
  __syncthreads();
  // GEMM-b: acc2 = inter @ dec^T  (A from LDS -> ds_load path)
  v8f acc2[8];
#pragma unroll
  for (int j = 0; j < 8; j++) acc2[j] = zero_v8f();
  const __bf16* Ap2 = &inter[(rg * 16) * 256];
#pragma unroll
  for (int k0 = 0; k0 < CCH; k0 += 32) {
    v16bf a = load_a16x32(Ap2 + k0, 256, lane);
#pragma unroll
    for (int j = 0; j < 8; j++) {
      v16bf b = load_b32x16(Dc16 + (size_t)(ch * 128 + j * 16) * CCH + k0, CCH, lane);
      acc2[j] = wmma_bf16(a, b, acc2[j]);
    }
  }
  // residual + transposed store, 32 rows at a time (fbuf reuses inter's LDS)
  for (int half = 0; half < 2; half++) {
    __syncthreads();  // first iter: all inter reads done before fbuf overwrite
    if ((rg >> 1) == half) {
#pragma unroll
      for (int j = 0; j < 8; j++)
#pragma unroll
        for (int r = 0; r < 8; r++) {
          int lr = rb + r, lc = cb0 + j * 16;
          float res = X32[(size_t)(row0 + lr) * CCH + lc];
          fbuf[(lr & 31) * 260 + lc] = acc2[j][r] + res;
        }
    }
    __syncthreads();
    int l = t & 31, cq = t >> 5;  // 8 groups x 32 cols
    for (int jj = 0; jj < 32; jj++) {
      int c = cq * 32 + jj;
      out[((size_t)(bb * CCH + c)) * LL + l0 + half * 32 + l] = fbuf[l * 260 + c];
    }
  }
}

extern "C" void kernel_launch(void* const* d_in, const int* in_sizes, int n_in,
                              void* d_out, int out_size, void* d_ws, size_t ws_size,
                              hipStream_t stream) {
  (void)in_sizes; (void)n_in; (void)out_size; (void)ws_size;
  const float* x = (const float*)d_in[0];
  const float* Lre = (const float*)d_in[1];
  const float* Lim = (const float*)d_in[2];
  const float* Bre = (const float*)d_in[3];
  const float* Bim = (const float*)d_in[4];
  const float* Cre = (const float*)d_in[5];
  const float* Cim = (const float*)d_in[6];
  const float* Dv = (const float*)d_in[7];
  const float* lstep = (const float*)d_in[8];
  const float* w1 = (const float*)d_in[9];
  const float* b1 = (const float*)d_in[10];
  const float* w2 = (const float*)d_in[11];
  const float* b2 = (const float*)d_in[12];
  const float* enc = (const float*)d_in[13];
  const float* dec = (const float*)d_in[14];
  float* out = (float*)d_out;

  char* base = (char*)d_ws;
  size_t off = 0;
  auto take = [&](size_t bytes) -> char* {
    char* p = base + off;
    off += (bytes + 255) & ~(size_t)255;
    return p;
  };
  float* fx32 = (float*)take((size_t)NROWS * CCH * 4);
  __bf16* fx16 = (__bf16*)take((size_t)NROWS * CCH * 2);
  float2* Bu = (float2*)take((size_t)NROWS * PP * 8);
  __bf16* xsre = (__bf16*)take((size_t)NROWS * PP * 2);
  __bf16* xsim = (__bf16*)take((size_t)NROWS * PP * 2);
  float2* carry = (float2*)take((size_t)NB * NCHUNK * PP * 8);
  float2* LamBar = (float2*)take(PP * 8);
  float2* Coef = (float2*)take(PP * 8);
  float2* LamPow = (float2*)take(PP * 8);
  __bf16* Bbre16 = (__bf16*)take((size_t)PP * CCH * 2);
  __bf16* Bbim16 = (__bf16*)take((size_t)PP * CCH * 2);
  __bf16* Cre16 = (__bf16*)take((size_t)CCH * PP * 2);
  __bf16* nCim16 = (__bf16*)take((size_t)CCH * PP * 2);
  __bf16* enc16 = (__bf16*)take((size_t)CCH * CCH * 2);
  __bf16* dec16 = (__bf16*)take((size_t)CCH * CCH * 2);
  // region reuse (kernel-order safe): Bu dead after scan3 -> fx2 f32;
  // fx16 dead after gemm1 -> fx2 bf16
  float* fx2f = (float*)Bu;
  __bf16* fx2h = fx16;

  k_prep1<<<1, 256, 0, stream>>>(Lre, Lim, lstep, LamBar, Coef, LamPow);
  k_prep2<<<256, 256, 0, stream>>>(Bre, Bim, Coef, Cre, Cim, enc, dec, Bbre16,
                                   Bbim16, Cre16, nCim16, enc16, dec16);
  k_ln1<<<NB * (LL / 64), 256, 0, stream>>>(x, w1, b1, fx32, fx16);
  k_gemm1<<<(NROWS / 16) * (PP / 64) / 8, 256, 0, stream>>>(fx16, Bbre16, Bbim16, Bu);
  k_scan1<<<(NB * NCHUNK * PP) / 256, 256, 0, stream>>>(Bu, LamBar, carry);
  k_scan2<<<(NB * PP) / 256, 256, 0, stream>>>(LamPow, carry);
  k_scan3<<<(NB * NCHUNK * PP) / 256, 256, 0, stream>>>(Bu, LamBar, carry, xsre, xsim);
  k_gemm2<<<(NROWS / 16) * (CCH / 64) / 8, 256, 0, stream>>>(xsre, xsim, Cre16,
                                                             nCim16, Dv, fx32);
  k_ln2<<<NROWS / 8, 256, 0, stream>>>(fx32, w2, b2, fx2f, fx2h);
  k_ff<<<NROWS / 64, 256, 0, stream>>>(fx2h, fx2f, enc16, dec16, out);
}
